// match_block_57767310131218
// MI455X (gfx1250) — matrix operand order repaired and checked by reading, not compiled, not run on hardware
//
#include <hip/hip_runtime.h>
#include <math.h>

// ---------------------------------------------------------------------------
// Problem constants (from reference): B=8, C=512, H=W=48 -> N=2304,
// inter=256, hid=32.  All GEMM dims divide the tile sizes exactly.
// ---------------------------------------------------------------------------
#define BATCH 8
#define CCH   512
#define NS    2304        // 48*48
#define INTER 256
#define HID   32
#define BN_EPS 1e-5f

typedef __attribute__((ext_vector_type(8)))  float  v8f;
typedef __attribute__((ext_vector_type(16))) __bf16 v16bf;

__device__ __forceinline__ __bf16 f2bf(float f) {
  unsigned u = __builtin_bit_cast(unsigned, f);
  unsigned r = (u + 0x7FFFu + ((u >> 16) & 1u)) >> 16;
  unsigned short s = (unsigned short)r;
  return __builtin_bit_cast(__bf16, s);
}

// Async DMA: 16 bytes per lane, global -> LDS, per-lane LDS dest address.
// Tracked by ASYNCcnt (cdna5_isa 15.18.3 / CDNA5 ISA).
__device__ __forceinline__ void async_g2l_b128(unsigned lds_addr, const void* gptr) {
  asm volatile("global_load_async_to_lds_b128 %0, %1, off"
               :: "v"(lds_addr), "v"((unsigned long long)gptr)
               : "memory");
}
__device__ __forceinline__ void wait_async0() {
  asm volatile("s_wait_asynccnt 0x0" ::: "memory");
}

// ---------------------------------------------------------------------------
// fp32 -> bf16 conversion (grid-stride), plain layout
// ---------------------------------------------------------------------------
__global__ void cvt_f32_bf16(const float* __restrict__ x, __bf16* __restrict__ y,
                             long long n) {
  long long i = (long long)blockIdx.x * blockDim.x + threadIdx.x;
  long long stride = (long long)gridDim.x * blockDim.x;
  for (; i < n; i += stride) y[i] = f2bf(x[i]);
}

// ---------------------------------------------------------------------------
// fp32 [B][C][N] -> bf16 transposed [B][N][C] (so channel becomes the unit-
// stride K axis for the front GEMMs).
// ---------------------------------------------------------------------------
__global__ void cvt_t_bf16(const float* __restrict__ x, __bf16* __restrict__ y) {
  const long long total = (long long)BATCH * CCH * NS;
  long long i = (long long)blockIdx.x * blockDim.x + threadIdx.x;
  long long stride = (long long)gridDim.x * blockDim.x;
  for (; i < total; i += stride) {
    const int c = (int)(i % CCH);
    const int n = (int)((i / CCH) % NS);
    const int b = (int)(i / ((long long)CCH * NS));
    y[i] = f2bf(x[((long long)b * CCH + c) * NS + n]);
  }
}

// ---------------------------------------------------------------------------
// Batched GEMM, bf16 in, fp32 accumulate via v_wmma_f32_16x16x32_bf16.
//   Out[m,n] = alpha * sum_k A[m,k]*B[k,n] + bias[m]
// Both operands are unit-K-stride:  A[m,k] = Aptr[m*sAm + k],
//                                   B[k,n] = Bptr[n*sBn + k].
// (The launcher arranges every tensor layout so this always holds.)
//
// Block tile 128(M) x 128(N), K-step 32.  256 threads = 8 waves (4 M x 2 N);
// each wave owns 32x64 = 2x4 WMMA 16x16x32 tiles (8 v_wmma per K-step).
//
// LDS is double-buffered and pre-swizzled at *store* time so fragment loads
// are contiguous 32-byte vector loads (2x ds_load_b128, no repacking):
//   lA: fragment order  [mtile(8)][lane(32)][e(16)]   (ISA 7.12.2 A layout)
//   lB: n-major         [n(128)][k(32)]               (slot k order 0-7,16-23,8-15,24-31; half reads slots half*16..)
// Staging is pure async DMA: each thread owns four loop-invariant 16-byte
// LDS slots and issues GLOBAL_LOAD_ASYNC_TO_LDS_B128 for the *next* K tile
// while the WMMA chain consumes the current one (ping-pong, one barrier per
// K-step, s_wait_asynccnt 0 before the barrier).
//
// Epilogue can emit: Obf  (bf16 row-major O[m*N+n]),
//                    ObfT (bf16 transposed O[n*M+m]),
//                    Of32 (fp32 row-major) — any subset non-null.
// Requires M%128==0, N%128==0, K%32==0.
// ---------------------------------------------------------------------------
#define BM 128
#define BN 128
#define BK 32

__global__ __launch_bounds__(256)
void gemm_bf16_wmma(const __bf16* __restrict__ A, long long sAm, long long sAb,
                    const __bf16* __restrict__ B, long long sBn, long long sBb,
                    const float* __restrict__ bias, float alpha,
                    __bf16* __restrict__ Obf, __bf16* __restrict__ ObfT,
                    float* __restrict__ Of32, long long sOb,
                    int M, int N, int K) {
  __shared__ __bf16 lA[2 * BM * BK];   // fragment-ordered, double-buffered
  __shared__ __bf16 lB[2 * BN * BK];   // n-major, double-buffered
  const unsigned BUFBYTES = BM * BK * 2;   // 8192 (same for lB)

  const int bz = blockIdx.z;
  A += (long long)bz * sAb;
  B += (long long)bz * sBb;
  const long long obase = (long long)bz * sOb;

  const int tid  = threadIdx.x;
  const int lane = tid & 31;
  const int wid  = tid >> 5;
  const int wm   = wid >> 1;       // 0..3  (M direction)
  const int wn   = wid & 1;        // 0..1  (N direction)
  const int half = lane >> 4;      // lane group 0/1
  const int ml   = lane & 15;

  const int M0 = blockIdx.y * BM;
  const int N0 = blockIdx.x * BN;

  // ---- per-thread staging descriptors (2 A + 2 B 16-byte chunks) ---------
  const __bf16 *pA0, *pA1, *pB0, *pB1;
  unsigned ldsA0, ldsA1, ldsB0, ldsB1;
  {
    // chunk 0: t = tid, chunk 1: t = tid + 256
    const int t0 = tid, t1 = tid + 256;
    // A chunks: row r = t>>2, k offset c0 = (t&3)*8
    {
      const int r = t0 >> 2, c0 = (t0 & 3) << 3;
      pA0 = A + (long long)(M0 + r) * sAm + c0;
      const int mtile = r >> 4, m = r & 15;
      ldsA0 = (unsigned)(unsigned long long)(void*)
              &lA[((mtile * 32 + m + 16 * ((c0 >> 3) & 1)) << 4) + ((c0 >> 4) << 3)];
    }
    {
      const int r = t1 >> 2, c0 = (t1 & 3) << 3;
      pA1 = A + (long long)(M0 + r) * sAm + c0;
      const int mtile = r >> 4, m = r & 15;
      ldsA1 = (unsigned)(unsigned long long)(void*)
              &lA[((mtile * 32 + m + 16 * ((c0 >> 3) & 1)) << 4) + ((c0 >> 4) << 3)];
    }
    // B chunks: col n = t>>2, k offset c0 = (t&3)*8
    {
      const int n = t0 >> 2, c0 = (t0 & 3) << 3;
      pB0 = B + (long long)(N0 + n) * sBn + c0;
      ldsB0 = (unsigned)(unsigned long long)(void*)&lB[(n << 5) + (((c0 >> 3) & 1) << 4) + ((c0 >> 4) << 3)];
    }
    {
      const int n = t1 >> 2, c0 = (t1 & 3) << 3;
      pB1 = B + (long long)(N0 + n) * sBn + c0;
      ldsB1 = (unsigned)(unsigned long long)(void*)&lB[(n << 5) + (((c0 >> 3) & 1) << 4) + ((c0 >> 4) << 3)];
    }
  }

  // ---- loop-invariant fragment LDS element indices -----------------------
  int ifA[2];
  int ifB[4];
#pragma unroll
  for (int mi = 0; mi < 2; ++mi)
    ifA[mi] = ((wm * 2 + mi) * 32 + lane) << 4;
#pragma unroll
  for (int ni = 0; ni < 4; ++ni)
    ifB[ni] = ((wn * 64 + ni * 16 + ml) << 5) + half * 16;

  v8f acc[2][4] = {};

  // ---- pipelined main loop ----------------------------------------------
  const int nK = K / BK;
  // prologue: tile 0 -> buffer 0
  async_g2l_b128(ldsA0, pA0); async_g2l_b128(ldsA1, pA1);
  async_g2l_b128(ldsB0, pB0); async_g2l_b128(ldsB1, pB1);
  pA0 += BK; pA1 += BK; pB0 += BK; pB1 += BK;
  wait_async0();
  __syncthreads();

  for (int i = 0; i < nK; ++i) {
    const int  cur  = i & 1;
    const bool more = (i + 1) < nK;
    if (more) {                         // issue tile i+1 into the other buffer
      const unsigned off = (unsigned)((i + 1) & 1) * BUFBYTES;
      async_g2l_b128(ldsA0 + off, pA0); async_g2l_b128(ldsA1 + off, pA1);
      async_g2l_b128(ldsB0 + off, pB0); async_g2l_b128(ldsB1 + off, pB1);
      pA0 += BK; pA1 += BK; pB0 += BK; pB1 += BK;
    }

    // compute tile i while the DMAs for tile i+1 are in flight
    const __bf16* bA = lA + cur * (BM * BK);
    const __bf16* bB = lB + cur * (BN * BK);
    v16bf afrag[2];
    v16bf bfrag[4];
#pragma unroll
    for (int mi = 0; mi < 2; ++mi) afrag[mi] = *(const v16bf*)(bA + ifA[mi]);
#pragma unroll
    for (int ni = 0; ni < 4; ++ni) bfrag[ni] = *(const v16bf*)(bB + ifB[ni]);

#pragma unroll
    for (int mi = 0; mi < 2; ++mi)
#pragma unroll
      for (int ni = 0; ni < 4; ++ni)
        acc[mi][ni] = __builtin_amdgcn_wmma_f32_16x16x32_bf16(
            false, afrag[mi], false, bfrag[ni], (short)0, acc[mi][ni], false, false);

    if (more) {
      wait_async0();                    // tile i+1 landed in LDS (this wave)
      __syncthreads();                  // ... and in all waves
    }
  }

  // ---- epilogue: D layout -> lane holds N=ml, VGPR r holds M=half*8+r ----
#pragma unroll
  for (int mi = 0; mi < 2; ++mi) {
#pragma unroll
    for (int ni = 0; ni < 4; ++ni) {
      const int ncol = N0 + wn * 64 + ni * 16 + ml;
#pragma unroll
      for (int r = 0; r < 8; ++r) {
        const int mrow = M0 + wm * 32 + mi * 16 + half * 8 + r;
        float v = alpha * acc[mi][ni][r];
        if (bias) v += bias[mrow];
        if (Of32) Of32[obase + (long long)mrow * N + ncol] = v;
        if (Obf)  Obf [obase + (long long)mrow * N + ncol] = f2bf(v);
        if (ObfT) ObfT[obase + (long long)ncol * M + mrow] = f2bf(v);
      }
    }
  }
}

// ---------------------------------------------------------------------------
// BN stats over (batch, spatial) per channel. x layout [B, C, N].
// ---------------------------------------------------------------------------
__global__ void bn_stats(const float* __restrict__ x, float* __restrict__ mean,
                         float* __restrict__ rstd) {
  const int c = blockIdx.x;
  const int tid = threadIdx.x;
  float s = 0.f, ss = 0.f;
  const int total = BATCH * NS;
  for (int i = tid; i < total; i += 256) {
    const int b = i / NS, n = i % NS;
    float v = x[((long long)b * CCH + c) * NS + n];
    s += v;
    ss += v * v;
  }
  __shared__ float rs[256], rq[256];
  rs[tid] = s; rq[tid] = ss;
  __syncthreads();
  for (int off = 128; off > 0; off >>= 1) {
    if (tid < off) { rs[tid] += rs[tid + off]; rq[tid] += rq[tid + off]; }
    __syncthreads();
  }
  if (tid == 0) {
    const float inv = 1.f / (float)total;
    const float m = rs[0] * inv;
    const float var = rq[0] * inv - m * m;   // biased variance (torch BN)
    mean[c] = m;
    rstd[c] = rsqrtf(var + BN_EPS);
  }
}

// ---------------------------------------------------------------------------
// In-place BN normalize + affine + residual.
// ---------------------------------------------------------------------------
__global__ void bn_apply_res(float* __restrict__ x, const float* __restrict__ res,
                             const float* __restrict__ mean, const float* __restrict__ rstd,
                             const float* __restrict__ gamma, const float* __restrict__ beta,
                             long long total) {
  long long i = (long long)blockIdx.x * blockDim.x + threadIdx.x;
  long long stride = (long long)gridDim.x * blockDim.x;
  for (; i < total; i += stride) {
    const int c = (int)((i / NS) % CCH);
    x[i] = (x[i] - mean[c]) * rstd[c] * gamma[c] + beta[c] + res[i];
  }
}

// ---------------------------------------------------------------------------
// Spatial avg + max pool per (b,c).
// ---------------------------------------------------------------------------
__global__ void pool_bc(const float* __restrict__ x, float* __restrict__ avgp,
                        float* __restrict__ maxp) {
  const long long base = (long long)blockIdx.x * NS;
  const int tid = threadIdx.x;
  float s = 0.f, mx = -3.4e38f;
  for (int n = tid; n < NS; n += 256) {
    float v = x[base + n];
    s += v;
    mx = fmaxf(mx, v);
  }
  __shared__ float rs[256], rm[256];
  rs[tid] = s; rm[tid] = mx;
  __syncthreads();
  for (int off = 128; off > 0; off >>= 1) {
    if (tid < off) { rs[tid] += rs[tid + off]; rm[tid] = fmaxf(rm[tid], rm[tid + off]); }
    __syncthreads();
  }
  if (tid == 0) {
    avgp[blockIdx.x] = rs[0] / (float)NS;
    maxp[blockIdx.x] = rm[0];
  }
}

// ---------------------------------------------------------------------------
// ChannelGate: scale[b,c] = sigmoid( mlp(avg)[c] + mlp(max)[c] )
// ---------------------------------------------------------------------------
__global__ void gate_mlp(const float* __restrict__ avgp, const float* __restrict__ maxp,
                         const float* __restrict__ w1, const float* __restrict__ b1,
                         const float* __restrict__ w2, const float* __restrict__ b2,
                         float* __restrict__ scale) {
  const int b = blockIdx.x;
  const float* pa = avgp + b * CCH;
  const float* pm = maxp + b * CCH;
  __shared__ float hA[HID], hM[HID];
  if (threadIdx.x < HID) {
    const int h = threadIdx.x;
    float sa = b1[h], sm = b1[h];
    for (int c = 0; c < CCH; ++c) {
      const float w = w1[h * CCH + c];
      sa += w * pa[c];
      sm += w * pm[c];
    }
    hA[h] = fmaxf(sa, 0.f);
    hM[h] = fmaxf(sm, 0.f);
  }
  __syncthreads();
  for (int c = threadIdx.x; c < CCH; c += blockDim.x) {
    float s = 2.f * b2[c];                 // +b2 from each of the two MLP passes
    for (int h = 0; h < HID; ++h) s += w2[c * HID + h] * (hA[h] + hM[h]);
    scale[b * CCH + c] = 1.f / (1.f + expf(-s));
  }
}

// ---------------------------------------------------------------------------
// In-place channel scaling: x[b,c,n] *= scale[i / NS]
// ---------------------------------------------------------------------------
__global__ void scale_apply(float* __restrict__ x, const float* __restrict__ scale,
                            long long total) {
  long long i = (long long)blockIdx.x * blockDim.x + threadIdx.x;
  long long stride = (long long)gridDim.x * blockDim.x;
  for (; i < total; i += stride) x[i] *= scale[i / NS];
}

// ---------------------------------------------------------------------------
// Host-side orchestration
// ---------------------------------------------------------------------------
static inline size_t align256(size_t x) { return (x + 255) & ~(size_t)255; }

extern "C" void kernel_launch(void* const* d_in, const int* in_sizes, int n_in,
                              void* d_out, int out_size, void* d_ws, size_t ws_size,
                              hipStream_t stream) {
  (void)in_sizes; (void)n_in; (void)out_size; (void)ws_size;

  const float* spt     = (const float*)d_in[0];
  const float* qry     = (const float*)d_in[1];
  const float* g_w     = (const float*)d_in[2];
  const float* g_b     = (const float*)d_in[3];
  const float* theta_w = (const float*)d_in[4];
  const float* theta_b = (const float*)d_in[5];
  const float* phi_w   = (const float*)d_in[6];
  const float* phi_b   = (const float*)d_in[7];
  const float* W_w     = (const float*)d_in[8];
  const float* W_b     = (const float*)d_in[9];
  const float* W_gamma = (const float*)d_in[10];
  const float* W_beta  = (const float*)d_in[11];
  const float* Q_w     = (const float*)d_in[12];
  const float* Q_b     = (const float*)d_in[13];
  const float* Q_gamma = (const float*)d_in[14];
  const float* Q_beta  = (const float*)d_in[15];
  const float* mlp_w1  = (const float*)d_in[16];
  const float* mlp_b1  = (const float*)d_in[17];
  const float* mlp_w2  = (const float*)d_in[18];
  const float* mlp_b2  = (const float*)d_in[19];

  const long long nAct  = (long long)BATCH * CCH * NS;       // 9,437,184
  const long long nInt  = (long long)BATCH * INTER * NS;     // 4,718,592
  const long long nF    = (long long)BATCH * NS * NS;        // 42,467,328
  const long long nWgtF = (long long)INTER * CCH;            // 131,072

  // Carve workspace
  char* p = (char*)d_ws;
  auto carve = [&](size_t bytes) { char* r = p; p += align256(bytes); return r; };
  __bf16* sptT_bf  = (__bf16*)carve(nAct * 2);   // [B][NS][C]
  __bf16* qryT_bf  = (__bf16*)carve(nAct * 2);   // [B][NS][C]
  __bf16* thetaT   = (__bf16*)carve(nInt * 2);   // [B][NS][INTER]
  __bf16* phiT     = (__bf16*)carve(nInt * 2);   // [B][NS][INTER]
  __bf16* gspt_bf  = (__bf16*)carve(nInt * 2);   // [B][INTER][NS]
  __bf16* gqry_bf  = (__bf16*)carve(nInt * 2);   // [B][INTER][NS]
  __bf16* naim_bf  = (__bf16*)carve(nInt * 2);   // [B][NS][INTER]
  __bf16* ndet_bf  = (__bf16*)carve(nInt * 2);   // [B][NS][INTER]
  __bf16* f_bf     = (__bf16*)carve(nF * 2);     // [B][Na][Nd]
  __bf16* fT_bf    = (__bf16*)carve(nF * 2);     // [B][Nd][Na]
  __bf16* gw_bf    = (__bf16*)carve(nWgtF * 2);
  __bf16* tw_bf    = (__bf16*)carve(nWgtF * 2);
  __bf16* pw_bf    = (__bf16*)carve(nWgtF * 2);
  __bf16* Ww_bf    = (__bf16*)carve(nWgtF * 2);
  __bf16* Qw_bf    = (__bf16*)carve(nWgtF * 2);
  float*  meanW    = (float*)carve(CCH * 4);
  float*  rstdW    = (float*)carve(CCH * 4);
  float*  meanQ    = (float*)carve(CCH * 4);
  float*  rstdQ    = (float*)carve(CCH * 4);
  float*  avgp     = (float*)carve(BATCH * CCH * 4);
  float*  maxp     = (float*)carve(BATCH * CCH * 4);
  float*  scl      = (float*)carve(BATCH * CCH * 4);

  float* out_det = (float*)d_out;            // act_det: [B, C, NS]
  float* out_aim = (float*)d_out + nAct;     // act_aim: [B, C, NS]

  // --- 1. convert: activations -> transposed bf16, weights -> bf16 ---
  cvt_t_bf16<<<4096, 256, 0, stream>>>(spt, sptT_bf);
  cvt_t_bf16<<<4096, 256, 0, stream>>>(qry, qryT_bf);
  cvt_f32_bf16<<<256, 256, 0, stream>>>(g_w,     gw_bf, nWgtF);
  cvt_f32_bf16<<<256, 256, 0, stream>>>(theta_w, tw_bf, nWgtF);
  cvt_f32_bf16<<<256, 256, 0, stream>>>(phi_w,   pw_bf, nWgtF);
  cvt_f32_bf16<<<256, 256, 0, stream>>>(W_w,     Ww_bf, nWgtF);
  cvt_f32_bf16<<<256, 256, 0, stream>>>(Q_w,     Qw_bf, nWgtF);

  const long long sAct = (long long)CCH * NS;    // batch stride, activations
  const long long sInt = (long long)INTER * NS;  // batch stride, inter maps
  const long long sF   = (long long)NS * NS;

  // --- 2. front 1x1 convs: [INTER x CCH] @ [CCH x NS] + bias, per batch ---
  // B[k=c][n=sp] = actT[sp*C + c]  (unit-K)
  {
    dim3 g(NS / BN, INTER / BM, BATCH);
    // theta(spt) -> thetaT[a][i]
    gemm_bf16_wmma<<<g, 256, 0, stream>>>(tw_bf, CCH, 0,
                                          sptT_bf, CCH, sAct,
                                          theta_b, 1.f,
                                          nullptr, thetaT, nullptr, sInt,
                                          INTER, NS, CCH);
    // phi(qry) -> phiT[d][i]
    gemm_bf16_wmma<<<g, 256, 0, stream>>>(pw_bf, CCH, 0,
                                          qryT_bf, CCH, sAct,
                                          phi_b, 1.f,
                                          nullptr, phiT, nullptr, sInt,
                                          INTER, NS, CCH);
    // g(spt) -> a_x^T, row-major [INTER][NS]
    gemm_bf16_wmma<<<g, 256, 0, stream>>>(gw_bf, CCH, 0,
                                          sptT_bf, CCH, sAct,
                                          g_b, 1.f,
                                          gspt_bf, nullptr, nullptr, sInt,
                                          INTER, NS, CCH);
    // g(qry) -> d_x^T, row-major [INTER][NS]
    gemm_bf16_wmma<<<g, 256, 0, stream>>>(gw_bf, CCH, 0,
                                          qryT_bf, CCH, sAct,
                                          g_b, 1.f,
                                          gqry_bf, nullptr, nullptr, sInt,
                                          INTER, NS, CCH);
  }

  // --- 3. f[a,d] = sum_i thetaT[a,i] * phiT[d,i]; emit f and fT ---
  {
    dim3 g(NS / BN, NS / BM, BATCH);
    gemm_bf16_wmma<<<g, 256, 0, stream>>>(thetaT, INTER, sInt,
                                          phiT, INTER, sInt,
                                          nullptr, 1.f,
                                          f_bf, fT_bf, nullptr, sF,
                                          NS, NS, INTER);
  }

  // --- 4. attention applies (alpha folds the /Nd, /Na) ---
  {
    dim3 g(NS / BN, INTER / BM, BATCH);
    const float inv = 1.f / (float)NS;
    // non_aim[i,a] = (1/Nd) sum_d gqry[i,d] * f[a,d];  B[k=d][n=a] = f[a*NS+d]
    gemm_bf16_wmma<<<g, 256, 0, stream>>>(gqry_bf, NS, sInt,
                                          f_bf, NS, sF,
                                          nullptr, inv,
                                          nullptr, naim_bf, nullptr, sInt,
                                          INTER, NS, NS);
    // non_det[i,d] = (1/Na) sum_a gspt[i,a] * f[a,d]; B[k=a][n=d] = fT[d*NS+a]
    gemm_bf16_wmma<<<g, 256, 0, stream>>>(gspt_bf, NS, sInt,
                                          fT_bf, NS, sF,
                                          nullptr, inv,
                                          nullptr, ndet_bf, nullptr, sInt,
                                          INTER, NS, NS);
  }

  // --- 5. back 1x1 convs into d_out (fp32, pre-BN) ---
  // B[k=i][n=sp] = naim/ndet[sp*INTER + i]  (unit-K)
  {
    dim3 g(NS / BN, CCH / BM, BATCH);
    gemm_bf16_wmma<<<g, 256, 0, stream>>>(Ww_bf, INTER, 0,
                                          naim_bf, INTER, sInt,
                                          W_b, 1.f,
                                          nullptr, nullptr, out_aim, sAct,
                                          CCH, NS, INTER);
    gemm_bf16_wmma<<<g, 256, 0, stream>>>(Qw_bf, INTER, 0,
                                          ndet_bf, INTER, sInt,
                                          Q_b, 1.f,
                                          nullptr, nullptr, out_det, sAct,
                                          CCH, NS, INTER);
  }

  // --- 6. BN (training-mode, biased var) + residual, in place ---
  bn_stats<<<CCH, 256, 0, stream>>>(out_aim, meanW, rstdW);
  bn_stats<<<CCH, 256, 0, stream>>>(out_det, meanQ, rstdQ);
  bn_apply_res<<<8192, 256, 0, stream>>>(out_aim, spt, meanW, rstdW, W_gamma, W_beta, nAct);
  bn_apply_res<<<8192, 256, 0, stream>>>(out_det, qry, meanQ, rstdQ, Q_gamma, Q_beta, nAct);

  // --- 7. ChannelGate on non_aim ---
  pool_bc<<<BATCH * CCH, 256, 0, stream>>>(out_aim, avgp, maxp);
  gate_mlp<<<BATCH, 256, 0, stream>>>(avgp, maxp, mlp_w1, mlp_b1, mlp_w2, mlp_b2, scl);
  scale_apply<<<8192, 256, 0, stream>>>(out_aim, scl, nAct);
  scale_apply<<<8192, 256, 0, stream>>>(out_det, scl, nAct);
}
